// SelfAttention_v2_6124623364479
// MI455X (gfx1250) — compile-verified, matrix operand order checked by reading, not run on hardware
//
#include <hip/hip_runtime.h>
#include <hip/hip_bf16.h>

// ---------------------------------------------------------------------------
// Self-attention, SEQ=4096, D=768, gfx1250.
// Round 5: QK^T split across d-half wave pairs (no duplicated score WMMAs),
// cross-wave softmax via LDS stats exchange, CK=64, async Q staging.
// ---------------------------------------------------------------------------

#define SEQ    4096
#define DMODEL  768
#define DHALF   384
#define CK       64      // keys per chunk in the attention kernel

typedef __attribute__((ext_vector_type(16))) _Float16 v16h;
typedef __attribute__((ext_vector_type(8)))  _Float16 v8h;
typedef __attribute__((ext_vector_type(4)))  _Float16 v4h;
typedef __attribute__((ext_vector_type(8)))  float    v8f;

#define WMMA_F16(a, b, c) \
  __builtin_amdgcn_wmma_f32_16x16x32_f16(false, (a), false, (b), (short)0, (c), false, false)

// Fragment gather for 16x32 A (or 32x16 B as transpose of row-major storage):
// lane: r = lane&15 -> row; kb = (lane>>4)*8; halves 0..7 = K[kb..kb+8),
// halves 8..15 = K[kb+16..kb+24).  Contiguous-K storage -> two 16B loads.
__device__ __forceinline__ v16h load_frag(const _Float16* __restrict__ base,
                                          int ld, int r, int kb) {
  const _Float16* p = base + (size_t)r * ld + kb;
  v8h lo = *(const v8h*)(p);
  v8h hi = *(const v8h*)(p + 16);
  v16h f;
#pragma unroll
  for (int i = 0; i < 8; ++i) { f[i] = lo[i]; f[i + 8] = hi[i]; }
  return f;
}

__device__ __forceinline__ v16h load_frag_lds(const _Float16* base,
                                              int ld, int r, int kb) {
  const _Float16* p = base + r * ld + kb;
  v8h lo = *(const v8h*)(p);
  v8h hi = *(const v8h*)(p + 16);
  v16h f;
#pragma unroll
  for (int i = 0; i < 8; ++i) { f[i] = lo[i]; f[i + 8] = hi[i]; }
  return f;
}

// CDNA5 async global->LDS copy, 16 bytes per lane (tracked by ASYNCcnt).
__device__ __forceinline__ void async_copy_b128(unsigned lds_off,
                                                const void* gaddr) {
  asm volatile("global_load_async_to_lds_b128 %0, %1, off"
               :: "v"(lds_off), "v"(gaddr) : "memory");
}
__device__ __forceinline__ void wait_asynccnt0() {
  asm volatile("s_wait_asynccnt 0x0" ::: "memory");
}

// ---------------------------------------------------------------------------
// fp32 -> fp16 conversion (vectorized x4)
// ---------------------------------------------------------------------------
__global__ void cvt_f32_f16(const float* __restrict__ in,
                            _Float16* __restrict__ out, int n4) {
  int i = blockIdx.x * blockDim.x + threadIdx.x;
  if (i < n4) {
    float4 v = ((const float4*)in)[i];
    v4h h;
    h[0] = (_Float16)v.x; h[1] = (_Float16)v.y;
    h[2] = (_Float16)v.z; h[3] = (_Float16)v.w;
    ((v4h*)out)[i] = h;
  }
}

// ---------------------------------------------------------------------------
// Projection.  z=0: Q[s,o], z=1: K[s,o], z=2: V stored TRANSPOSED as Vt[o,s].
// Block: 256 threads = 8 waves.  Wave tile 64x64, ping-pong k pipeline.
// Grid: (SEQ/128, DMODEL/256, 3).
// ---------------------------------------------------------------------------
__global__ __launch_bounds__(256) void proj_kernel(
    const _Float16* __restrict__ X,      // [SEQ, DMODEL]
    const _Float16* __restrict__ W,      // [3, DMODEL, DMODEL]
    _Float16* __restrict__ Y,            // Q,K: [2, SEQ, DMODEL]
    _Float16* __restrict__ Yt) {         // Vt:  [DMODEL, SEQ]
  const _Float16* Wb = W + (size_t)blockIdx.z * DMODEL * DMODEL;

  const int wave = threadIdx.x >> 5;
  const int lane = threadIdx.x & 31;
  const int r  = lane & 15;
  const int kb = (lane >> 4) * 8;
  const int h  = lane >> 4;

  const int srow0 = blockIdx.x * 128 + (wave >> 2) * 64;
  const int o0    = blockIdx.y * 256 + (wave & 3) * 64;
  const _Float16* Xb = X + (size_t)srow0 * DMODEL;

  v8f acc[4][4] = {};
  v16h a0[4], b0[4], a1[4], b1[4];
#pragma unroll
  for (int i = 0; i < 4; ++i) {
    a0[i] = load_frag(Xb + (size_t)(i * 16) * DMODEL, DMODEL, r, kb);
    b0[i] = load_frag(Wb + (size_t)(o0 + i * 16) * DMODEL, DMODEL, r, kb);
  }

  for (int kd = 0; kd < DMODEL; kd += 64) {
#pragma unroll
    for (int i = 0; i < 4; ++i) {
      a1[i] = load_frag(Xb + (size_t)(i * 16) * DMODEL + kd + 32, DMODEL, r, kb);
      b1[i] = load_frag(Wb + (size_t)(o0 + i * 16) * DMODEL + kd + 32, DMODEL, r, kb);
    }
#pragma unroll
    for (int i = 0; i < 4; ++i)
#pragma unroll
      for (int j = 0; j < 4; ++j)
        acc[i][j] = WMMA_F16(a0[i], b0[j], acc[i][j]);

    const int kd2 = (kd + 64 < DMODEL) ? (kd + 64) : 0;
#pragma unroll
    for (int i = 0; i < 4; ++i) {
      a0[i] = load_frag(Xb + (size_t)(i * 16) * DMODEL + kd2, DMODEL, r, kb);
      b0[i] = load_frag(Wb + (size_t)(o0 + i * 16) * DMODEL + kd2, DMODEL, r, kb);
    }
#pragma unroll
    for (int i = 0; i < 4; ++i)
#pragma unroll
      for (int j = 0; j < 4; ++j)
        acc[i][j] = WMMA_F16(a1[i], b1[j], acc[i][j]);
  }

  if (blockIdx.z == 2) {
    // V: store transposed  Vt[o][s]
#pragma unroll
    for (int i = 0; i < 4; ++i)
#pragma unroll
      for (int j = 0; j < 4; ++j)
#pragma unroll
        for (int rr = 0; rr < 8; ++rr) {
          int row = srow0 + i * 16 + rr + 8 * h;   // s
          int col = o0 + j * 16 + r;               // o
          Yt[(size_t)col * SEQ + row] = (_Float16)acc[i][j][rr];
        }
  } else {
    _Float16* Yb = Y + (size_t)blockIdx.z * SEQ * DMODEL;
#pragma unroll
    for (int i = 0; i < 4; ++i)
#pragma unroll
      for (int j = 0; j < 4; ++j)
#pragma unroll
        for (int rr = 0; rr < 8; ++rr) {
          int row = srow0 + i * 16 + rr + 8 * h;
          int col = o0 + j * 16 + r;
          Yb[(size_t)row * DMODEL + col] = (_Float16)acc[i][j][rr];
        }
  }
}

// ---------------------------------------------------------------------------
// Fused flash attention, CK=64, cooperative score computation.
// Block: 128 threads = 4 waves = 2 row-groups (16 Q rows) x 2 d-halves (384).
// The two d-half waves of a row-group each compute HALF the score tiles
// (32 keys) and exchange max/sum/P through LDS -> no duplicated QK^T WMMAs.
// ---------------------------------------------------------------------------
__global__ __launch_bounds__(128) void attn_kernel(
    const _Float16* __restrict__ Q,    // [SEQ, DMODEL]
    const _Float16* __restrict__ K,    // [SEQ, DMODEL]
    const _Float16* __restrict__ Vt,   // [DMODEL, SEQ]  (transposed V)
    float* __restrict__ out) {         // [SEQ, DMODEL] f32
  __shared__ __attribute__((aligned(16))) _Float16 Qs[32 * DMODEL];  // 48KB
  __shared__ __attribute__((aligned(16))) _Float16 Pr[2][16 * CK];   // 4KB
  __shared__ float Ml[2][2][16];   // per row-group, per d-half: row maxes
  __shared__ float Sl[2][2][16];   // per row-group, per d-half: row sums

  const int tid  = threadIdx.x;
  const int wave = tid >> 5;
  const int lane = tid & 31;
  const int rg = wave >> 1;     // row group 0..1
  const int dh = wave & 1;      // d-half 0..1
  const int r  = lane & 15;
  const int kb = (lane >> 4) * 8;
  const int h  = lane >> 4;

  const int d0 = dh * DHALF;
  const float scale = 0.03608439182435161f;   // 1/sqrt(768)

  // Async-stage this block's 32 Q rows into LDS (ASYNCcnt-tracked DMA path).
  const _Float16* Qblk = Q + (size_t)(blockIdx.x * 32) * DMODEL;
  for (int e = tid * 8; e < 32 * DMODEL; e += 128 * 8)
    async_copy_b128((unsigned)(uintptr_t)(Qs + e), Qblk + e);
  wait_asynccnt0();
  __syncthreads();

  const _Float16* Qw = Qs + rg * 16 * DMODEL;   // this wave's 16 rows

  v8f acc[24] = {};
  float mrun[8], lrun[8];
#pragma unroll
  for (int i = 0; i < 8; ++i) { mrun[i] = -1e30f; lrun[i] = 0.0f; }

  for (int kc = 0; kc < SEQ; kc += CK) {
    // Prefetch next chunk's K rows and Vt row segments toward L2/WGP$.
    {
      const int kcn = (kc + CK < SEQ) ? (kc + CK) : kc;
      const _Float16* Kn = K + (size_t)kcn * DMODEL;
      for (int e = tid * 64; e < CK * DMODEL; e += 128 * 64)
        __builtin_prefetch(Kn + e, 0, 0);
      for (int d = tid; d < DMODEL; d += 128)
        __builtin_prefetch(Vt + (size_t)d * SEQ + kcn, 0, 0);
    }

    // ---- This wave's half of the scores: keys [kc+dh*32, kc+dh*32+32) ----
    const _Float16* Kc = K + (size_t)(kc + dh * 32) * DMODEL;
    v16h ku[2], kv[2];
#pragma unroll
    for (int n = 0; n < 2; ++n)
      ku[n] = load_frag(Kc + (size_t)(n * 16) * DMODEL, DMODEL, r, kb);
    v8f s[2] = {};
    for (int kd = 0; kd < DMODEL; kd += 64) {
#pragma unroll
      for (int n = 0; n < 2; ++n)
        kv[n] = load_frag(Kc + (size_t)(n * 16) * DMODEL + kd + 32, DMODEL, r, kb);
      v16h aq0 = load_frag_lds(Qw + kd, DMODEL, r, kb);
#pragma unroll
      for (int n = 0; n < 2; ++n) s[n] = WMMA_F16(aq0, ku[n], s[n]);

      const int kd2 = (kd + 64 < DMODEL) ? (kd + 64) : 0;
#pragma unroll
      for (int n = 0; n < 2; ++n)
        ku[n] = load_frag(Kc + (size_t)(n * 16) * DMODEL + kd2, DMODEL, r, kb);
      v16h aq1 = load_frag_lds(Qw + kd + 32, DMODEL, r, kb);
#pragma unroll
      for (int n = 0; n < 2; ++n) s[n] = WMMA_F16(aq1, kv[n], s[n]);
    }

    // ---- partial row max over this wave's 32 keys; publish to LDS ----
    float pmax[8];
#pragma unroll
    for (int rr = 0; rr < 8; ++rr) {
      float t = fmaxf(s[0][rr] * scale, s[1][rr] * scale);
#pragma unroll
      for (int off = 8; off >= 1; off >>= 1)
        t = fmaxf(t, __shfl_xor(t, off, 32));
      pmax[rr] = t;
      if (r == 0) Ml[rg][dh][rr + 8 * h] = t;
    }
    __syncthreads();   // barrier A: row maxes visible; prev-chunk P reads done

    // ---- combined max, exponentials, partial sums, write P half ----
    float sf[8], psum[8];
#pragma unroll
    for (int rr = 0; rr < 8; ++rr) {
      float om = Ml[rg][1 - dh][rr + 8 * h];
      float mn = fmaxf(mrun[rr], fmaxf(pmax[rr], om));
      float sc = __expf(mrun[rr] - mn);
      mrun[rr] = mn;
      sf[rr] = sc;
      float p0 = __expf(s[0][rr] * scale - mn);
      float p1 = __expf(s[1][rr] * scale - mn);
      float rs = p0 + p1;
#pragma unroll
      for (int off = 8; off >= 1; off >>= 1)
        rs += __shfl_xor(rs, off, 32);
      psum[rr] = rs;
      if (r == 0) Sl[rg][dh][rr + 8 * h] = rs;
      int prow = rr + 8 * h;
      Pr[rg][prow * CK + dh * 32 + r]      = (_Float16)p0;
      Pr[rg][prow * CK + dh * 32 + 16 + r] = (_Float16)p1;
    }

    // rescale running accumulator by exp(m_old - m_new)
#pragma unroll
    for (int t = 0; t < 24; ++t)
#pragma unroll
      for (int rr = 0; rr < 8; ++rr) acc[t][rr] *= sf[rr];

    __syncthreads();   // barrier B: full P tile + partner sums visible

#pragma unroll
    for (int rr = 0; rr < 8; ++rr)
      lrun[rr] = lrun[rr] * sf[rr] + psum[rr] + Sl[rg][1 - dh][rr + 8 * h];

    // ---- O += P(16x64) @ V_chunk, B-frags from global V^T, ping-pong ----
    v16h aP0 = load_frag_lds(&Pr[rg][0], CK, r, kb);    // keys kc..kc+31
    v16h aP1 = load_frag_lds(&Pr[rg][32], CK, r, kb);   // keys kc+32..kc+63
    v16h va = load_frag(Vt + (size_t)d0 * SEQ + kc, SEQ, r, kb);
    v16h vb = load_frag(Vt + (size_t)d0 * SEQ + kc + 32, SEQ, r, kb);
#pragma unroll
    for (int t = 0; t < 24; ++t) {
      const int tn = (t + 1 < 24) ? (t + 1) : 0;
      v16h na = load_frag(Vt + (size_t)(d0 + tn * 16) * SEQ + kc, SEQ, r, kb);
      v16h nb = load_frag(Vt + (size_t)(d0 + tn * 16) * SEQ + kc + 32, SEQ, r, kb);
      acc[t] = WMMA_F16(aP0, va, acc[t]);
      acc[t] = WMMA_F16(aP1, vb, acc[t]);
      va = na; vb = nb;
    }
  }

  // ---- normalize and write f32 output ----
  const int q0 = blockIdx.x * 32 + rg * 16;
#pragma unroll
  for (int rr = 0; rr < 8; ++rr) {
    float inv = 1.0f / lrun[rr];
    int row = q0 + rr + 8 * h;
#pragma unroll
    for (int t = 0; t < 24; ++t)
      out[(size_t)row * DMODEL + d0 + t * 16 + r] = acc[t][rr] * inv;
  }
}

// ---------------------------------------------------------------------------
// Host launch
// ---------------------------------------------------------------------------
extern "C" void kernel_launch(void* const* d_in, const int* in_sizes, int n_in,
                              void* d_out, int out_size, void* d_ws, size_t ws_size,
                              hipStream_t stream) {
  const float* x  = (const float*)d_in[0];
  const float* Wq = (const float*)d_in[1];
  const float* Wk = (const float*)d_in[2];
  const float* Wv = (const float*)d_in[3];
  float* out = (float*)d_out;

  _Float16* xh  = (_Float16*)d_ws;                       // SEQ*DMODEL
  _Float16* wh  = xh + (size_t)SEQ * DMODEL;             // 3*DMODEL*DMODEL
  _Float16* qkv = wh + (size_t)3 * DMODEL * DMODEL;      // Q,K then Vt
  _Float16* vt  = qkv + (size_t)2 * SEQ * DMODEL;        // [DMODEL, SEQ]

  const int nX = SEQ * DMODEL;
  const int nW = DMODEL * DMODEL;

  cvt_f32_f16<<<(nX / 4 + 255) / 256, 256, 0, stream>>>(x, xh, nX / 4);
  cvt_f32_f16<<<(nW / 4 + 255) / 256, 256, 0, stream>>>(Wq, wh, nW / 4);
  cvt_f32_f16<<<(nW / 4 + 255) / 256, 256, 0, stream>>>(Wk, wh + (size_t)nW, nW / 4);
  cvt_f32_f16<<<(nW / 4 + 255) / 256, 256, 0, stream>>>(Wv, wh + (size_t)2 * nW, nW / 4);

  dim3 pgrid(SEQ / 128, DMODEL / 256, 3);
  proj_kernel<<<pgrid, 256, 0, stream>>>(xh, wh, qkv, vt);

  const _Float16* Qh = qkv;
  const _Float16* Kh = qkv + (size_t)SEQ * DMODEL;
  attn_kernel<<<SEQ / 32, 128, 0, stream>>>(Qh, Kh, vt, out);
}
